// Cell_3934190043855
// MI455X (gfx1250) — compile-verified, matrix operand order checked by reading, not run on hardware
//
#include <hip/hip_runtime.h>
#include <hip/hip_bf16.h>

#define N_NODES 50000
#define N_EDGES 800000
#define N_HID 256

typedef __attribute__((ext_vector_type(16))) __bf16 v16bf;
typedef __attribute__((ext_vector_type(8)))  float  v8f;

// ---------------------------------------------------------------------------
// GEMM: H[50000,256] = X[50000,256] @ W[256,256]^T + b, fp32 via bf16x3 WMMA.
// One wave per 16x16 output tile. 3125 M-tiles x 16 N-tiles = 50000 waves.
// Two independent accumulator chains (main = hi*hi, corr = lo*hi + hi*lo)
// so consecutive v_wmma ops are not serialized by the D->C dependency.
// A-frag layout (16-bit A 16x32): lane L holds row m = L&15, half = L>>4;
//   VGPR v in 0..3 -> K = half*8 + 2v (+1); v in 4..7 -> K = 16 + half*8 + 2(v-4).
// B-frag mirrors A with N in place of M; B[k][n] = W[n][k] (row of W, contiguous k).
// C/D layout: lane L -> n = L&15; VGPR r -> m = r + (L>>4)*8.
// ---------------------------------------------------------------------------
__global__ __launch_bounds__(256) void gemm_bf16x3_wmma(
    const float* __restrict__ X, const float* __restrict__ W,
    const float* __restrict__ bias, float* __restrict__ H) {
  const int lane = threadIdx.x & 31;
  const int wave = blockIdx.x * (blockDim.x >> 5) + (threadIdx.x >> 5);
  const int m0 = (wave >> 4) << 4;   // 0..49984
  const int n0 = (wave & 15) << 4;   // 0..240
  const int half = lane >> 4;
  const int sub  = lane & 15;

  const float* __restrict__ xrow = X + (size_t)(m0 + sub) * N_HID;
  const float* __restrict__ wrow = W + (size_t)(n0 + sub) * N_HID;

  v8f c_main = {};
  v8f c_corr = {};
  #pragma unroll
  for (int kc = 0; kc < 8; ++kc) {
    const int k0 = kc * 32;
    v16bf a_hi, a_lo, b_hi, b_lo;
    #pragma unroll
    for (int v = 0; v < 8; ++v) {
      const int kk = (v < 4) ? (half * 8 + 2 * v) : (16 + half * 8 + 2 * (v - 4));
      const float2 xa = *(const float2*)(xrow + k0 + kk);
      const float2 wb = *(const float2*)(wrow + k0 + kk);
      const __bf16 ax = (__bf16)xa.x, ay = (__bf16)xa.y;
      const __bf16 bx = (__bf16)wb.x, by = (__bf16)wb.y;
      a_hi[2 * v] = ax;                              a_hi[2 * v + 1] = ay;
      a_lo[2 * v] = (__bf16)(xa.x - (float)ax);      a_lo[2 * v + 1] = (__bf16)(xa.y - (float)ay);
      b_hi[2 * v] = bx;                              b_hi[2 * v + 1] = by;
      b_lo[2 * v] = (__bf16)(wb.x - (float)bx);      b_lo[2 * v + 1] = (__bf16)(wb.y - (float)by);
    }
    // independent chains: scheduler can interleave main/corr WMMAs
    c_corr = __builtin_amdgcn_wmma_f32_16x16x32_bf16(false, a_lo, false, b_hi, (short)0, c_corr, false, false);
    c_main = __builtin_amdgcn_wmma_f32_16x16x32_bf16(false, a_hi, false, b_hi, (short)0, c_main, false, false);
    c_corr = __builtin_amdgcn_wmma_f32_16x16x32_bf16(false, a_hi, false, b_lo, (short)0, c_corr, false, false);
  }

  const int n_out = n0 + sub;
  const float bn = bias[n_out];
  #pragma unroll
  for (int r = 0; r < 8; ++r) {
    const int m_out = m0 + r + half * 8;
    H[(size_t)m_out * N_HID + n_out] = (c_main[r] + c_corr[r]) + bn;
  }
}

// ---------------------------------------------------------------------------
// SPMM scatter: Y[rows[e]] += vals[e] * X[cols[e]], 256-wide rows.
// One wave per edge; lane does two float4 gathers (L2-resident state) and
// 8 hardware f32 atomic adds. Adjacency selected on-device via idx_arr[pos].
// ---------------------------------------------------------------------------
__global__ __launch_bounds__(256) void spmm_scatter(
    const int* __restrict__ rows_all, const int* __restrict__ cols_all,
    const float* __restrict__ vals_all, const int* __restrict__ idx_arr, int idx_pos,
    const float* __restrict__ Xin, float* __restrict__ Yout) {
  const int lane = threadIdx.x & 31;
  const long long e = (long long)blockIdx.x * (blockDim.x >> 5) + (threadIdx.x >> 5);
  if (e >= N_EDGES) return;
  const int a = idx_arr[idx_pos];
  const size_t base = (size_t)a * N_EDGES + (size_t)e;
  const int   r = rows_all[base];
  const int   cc = cols_all[base];
  const float v = vals_all[base];

  const float4 x0 = *(const float4*)(Xin + (size_t)cc * N_HID + lane * 4);
  const float4 x1 = *(const float4*)(Xin + (size_t)cc * N_HID + 128 + lane * 4);
  float* y0 = Yout + (size_t)r * N_HID + lane * 4;
  float* y1 = y0 + 128;
  unsafeAtomicAdd(y0 + 0, v * x0.x);
  unsafeAtomicAdd(y0 + 1, v * x0.y);
  unsafeAtomicAdd(y0 + 2, v * x0.z);
  unsafeAtomicAdd(y0 + 3, v * x0.w);
  unsafeAtomicAdd(y1 + 0, v * x1.x);
  unsafeAtomicAdd(y1 + 1, v * x1.y);
  unsafeAtomicAdd(y1 + 2, v * x1.z);
  unsafeAtomicAdd(y1 + 3, v * x1.w);
}

// ---------------------------------------------------------------------------
// LayerNorm (eps=1e-5) + exact-erf GELU. One wave per 256-wide row.
// ---------------------------------------------------------------------------
__global__ __launch_bounds__(256) void ln_gelu(
    const float* __restrict__ Xin, const float* __restrict__ gamma,
    const float* __restrict__ beta, float* __restrict__ Out) {
  const int lane = threadIdx.x & 31;
  const int row = blockIdx.x * (blockDim.x >> 5) + (threadIdx.x >> 5);
  const float* xr = Xin + (size_t)row * N_HID;
  const float4 x0 = *(const float4*)(xr + lane * 4);
  const float4 x1 = *(const float4*)(xr + 128 + lane * 4);

  float s  = x0.x + x0.y + x0.z + x0.w + x1.x + x1.y + x1.z + x1.w;
  float sq = x0.x * x0.x + x0.y * x0.y + x0.z * x0.z + x0.w * x0.w +
             x1.x * x1.x + x1.y * x1.y + x1.z * x1.z + x1.w * x1.w;
  #pragma unroll
  for (int m = 16; m >= 1; m >>= 1) {
    s  += __shfl_xor(s, m, 32);
    sq += __shfl_xor(sq, m, 32);
  }
  const float mean = s * (1.0f / N_HID);
  const float var  = sq * (1.0f / N_HID) - mean * mean;
  const float rstd = rsqrtf(var + 1e-5f);

  float* outr = Out + (size_t)row * N_HID;
  #pragma unroll
  for (int p = 0; p < 2; ++p) {
    const float4 xv = p ? x1 : x0;
    const int idx = lane * 4 + p * 128;
    const float4 g = *(const float4*)(gamma + idx);
    const float4 bb = *(const float4*)(beta + idx);
    float vv[4]  = {xv.x, xv.y, xv.z, xv.w};
    float gg[4]  = {g.x, g.y, g.z, g.w};
    float bbv[4] = {bb.x, bb.y, bb.z, bb.w};
    #pragma unroll
    for (int q = 0; q < 4; ++q) {
      const float y = (vv[q] - mean) * rstd * gg[q] + bbv[q];
      outr[idx + q] = 0.5f * y * (1.0f + erff(y * 0.70710678118654752f));
    }
  }
}

// ---------------------------------------------------------------------------
// Orchestration: s0 = GEMM; s1 = A[seq0]s0; s2 = A[seq1]s1 + A[res0]s0;
//                s3 = A[seq2]s2 + A[res1]s0 + A[res2]s1; out = GELU(LN(s3)).
// ---------------------------------------------------------------------------
extern "C" void kernel_launch(void* const* d_in, const int* in_sizes, int n_in,
                              void* d_out, int out_size, void* d_ws, size_t ws_size,
                              hipStream_t stream) {
  const float* x        = (const float*)d_in[0];
  const int*   adj_rows = (const int*)d_in[1];
  const int*   adj_cols = (const int*)d_in[2];
  const float* adj_vals = (const float*)d_in[3];
  const int*   idx_seq  = (const int*)d_in[4];
  const int*   idx_res  = (const int*)d_in[5];
  const float* W        = (const float*)d_in[6];
  const float* b        = (const float*)d_in[7];
  const float* gamma    = (const float*)d_in[8];
  const float* beta     = (const float*)d_in[9];
  float*       out      = (float*)d_out;

  const size_t SELEMS = (size_t)N_NODES * N_HID;
  const size_t SBYTES = SELEMS * sizeof(float);
  float* s0 = (float*)d_ws;
  float* s1 = s0 + SELEMS;
  float* s2 = s1 + SELEMS;
  float* s3 = s2 + SELEMS;

  const int gemm_blocks = (N_NODES / 16) * 16 / 8;  // 50000 waves / 8 waves-per-block
  const int spmm_blocks = N_EDGES / 8;              // 1 wave per edge
  const int row_blocks  = N_NODES / 8;              // 1 wave per row

  gemm_bf16x3_wmma<<<gemm_blocks, 256, 0, stream>>>(x, W, b, s0);

  hipMemsetAsync(s1, 0, SBYTES, stream);
  spmm_scatter<<<spmm_blocks, 256, 0, stream>>>(adj_rows, adj_cols, adj_vals, idx_seq, 0, s0, s1);

  hipMemsetAsync(s2, 0, SBYTES, stream);
  spmm_scatter<<<spmm_blocks, 256, 0, stream>>>(adj_rows, adj_cols, adj_vals, idx_seq, 1, s1, s2);
  spmm_scatter<<<spmm_blocks, 256, 0, stream>>>(adj_rows, adj_cols, adj_vals, idx_res, 0, s0, s2);

  hipMemsetAsync(s3, 0, SBYTES, stream);
  spmm_scatter<<<spmm_blocks, 256, 0, stream>>>(adj_rows, adj_cols, adj_vals, idx_seq, 2, s2, s3);
  spmm_scatter<<<spmm_blocks, 256, 0, stream>>>(adj_rows, adj_cols, adj_vals, idx_res, 1, s0, s3);
  spmm_scatter<<<spmm_blocks, 256, 0, stream>>>(adj_rows, adj_cols, adj_vals, idx_res, 2, s1, s3);

  ln_gelu<<<row_blocks, 256, 0, stream>>>(s3, gamma, beta, out);
}
